// GAE_11158325035213
// MI455X (gfx1250) — compile-verified, matrix operand order checked by reading, not run on hardware
//
#include <hip/hip_runtime.h>
#include <hip/hip_bf16.h>

// ---------------------------------------------------------------------------
// GAE forward for MI455X (gfx1250, wave32, WMMA 16x16x32 bf16, fp32 accum)
//   B1t  = (X @ W1)^T            bf16 [128][8192]
//   H    = relu(A @ B1)          bf16 [8192][128]   (row-major)
//   C2t  = (H @ W2)^T            bf16 [32][8192]
//   Z    = A @ C2                bf16 [8192][32]    (row-major)
//   out  = sigmoid(Z @ Z^T)      f32  [8192][8192]
//
// The two A_tilde passes stream 256MB fp32 each. Split-K x4 (2048 waves) +
// a 4-deep async-to-LDS ring per wave (global_load_async_to_lds_b128 /
// s_wait_asynccnt) keeps ~16MB of HBM traffic in flight — the bandwidth-delay
// product of 23.3 TB/s * ~700ns.
// ---------------------------------------------------------------------------

typedef __attribute__((ext_vector_type(16))) __bf16 bf16x16;
typedef __attribute__((ext_vector_type(8)))  __bf16 bf16x8;
typedef __attribute__((ext_vector_type(8)))  float  v8f;

#define GAE_N    8192
#define GAE_FIN  256
#define GAE_HID  128
#define GAE_LAT  32

static __device__ inline v8f zero8() {
    v8f v;
#pragma unroll
    for (int i = 0; i < 8; ++i) v[i] = 0.0f;
    return v;
}

static __device__ inline bf16x16 cat8(bf16x8 lo, bf16x8 hi) {
    return __builtin_shufflevector(lo, hi, 0, 1, 2, 3, 4, 5, 6, 7,
                                   8, 9, 10, 11, 12, 13, 14, 15);
}

// Pack 16 fp32 (two groups of 8) into the WMMA bf16 A/B per-lane layout.
static __device__ inline bf16x16 pack_bf16(float4 f0, float4 f1, float4 f2, float4 f3) {
    bf16x16 a;
    a[0] = (__bf16)f0.x;  a[1] = (__bf16)f0.y;  a[2] = (__bf16)f0.z;  a[3] = (__bf16)f0.w;
    a[4] = (__bf16)f1.x;  a[5] = (__bf16)f1.y;  a[6] = (__bf16)f1.z;  a[7] = (__bf16)f1.w;
    a[8] = (__bf16)f2.x;  a[9] = (__bf16)f2.y;  a[10] = (__bf16)f2.z; a[11] = (__bf16)f2.w;
    a[12] = (__bf16)f3.x; a[13] = (__bf16)f3.y; a[14] = (__bf16)f3.z; a[15] = (__bf16)f3.w;
    return a;
}

// 16-element bf16 operand load as two 8-runs at k0+8*half and k0+16+8*half (A from bf16 row-major, or
// B-operand from a K-contiguous transposed buffer Bt[col][K]).
static __device__ inline bf16x16 load_bf16_row(const __bf16* __restrict__ P,
                                               size_t ld, int row, int k0, int half) {
    const __bf16* p = P + (size_t)row * ld + (size_t)k0 + (size_t)half * 8;
    bf16x8 lo = *(const bf16x8*)(p);
    bf16x8 hi = *(const bf16x8*)(p + 16);
    return cat8(lo, hi);
}

static __device__ inline v8f wmma_bf16(bf16x16 a, bf16x16 b, v8f c) {
    return __builtin_amdgcn_wmma_f32_16x16x32_bf16(
        /*neg_a=*/false, a, /*neg_b=*/false, b,
        /*c_mod=*/(short)0, c, /*reuse_a=*/false, /*reuse_b=*/false);
}

// ---------------------------------------------------------------------------
// Split-K streaming GEMM:  C = op(A[MxK] * B[KxN]),  N = NT*16.
// Block = 128 threads = 4 waves on one 16-row strip; wave w owns K-quarter.
// fp32-A path: A tiles staged through a 4-deep per-wave LDS ring with
// global_load_async_to_lds_b128 (ASYNCcnt), consumed via ds_load_b128 with a
// 144B row pitch (conflict-minimal across 64 banks). B tiles (L2-resident)
// are register double-buffered. Partials reduced via LDS; epilogue:
//   RELU  : relu
//   TROUT : store transposed bf16 [N][M] (16B store/lane/tile) else [M][N].
// Requires K % 128 == 0, grid.x == M/16.
// ---------------------------------------------------------------------------
template <int NT, bool ABF16, bool RELU, bool TROUT>
__global__ __launch_bounds__(128) void gemm_splitk(const void* __restrict__ Aop,
                                                   const __bf16* __restrict__ Bt,
                                                   __bf16* __restrict__ Cout,
                                                   int M, int K) {
    constexpr int SPLIT = 4;
    const int lane = threadIdx.x & 31;
    const int half = lane >> 4;
    const int l    = lane & 15;
    const int wave = threadIdx.x >> 5;
    const int row0 = blockIdx.x * 16;

    const int Kw  = K / SPLIT;
    const int kb  = wave * Kw;          // this wave's K range start
    const int nIt = Kw / 32;

    v8f acc[NT];
#pragma unroll
    for (int t = 0; t < NT; ++t) acc[t] = zero8();

    bf16x16 b_cur[NT], b_nxt[NT];
#pragma unroll
    for (int t = 0; t < NT; ++t)
        b_cur[t] = load_bf16_row(Bt, (size_t)K, t * 16 + l, kb, half);

    if constexpr (!ABF16) {
        // ---------------- async-to-LDS pipelined fp32-A path ----------------
        constexpr int DEPTH  = 4;        // ring depth (groups in flight)
        constexpr int PITCHF = 36;       // floats per row (144B, bank-friendly)
        constexpr int SLOTF  = 16 * PITCHF;  // 576 floats = 2304B per slot
        __shared__ float ring[SPLIT][DEPTH][SLOTF];

        // Per-lane async-copy addressing: 128 16B-chunks per 16x32-f32 tile,
        // chunk = j*32 + lane, row = chunk>>3, col16 = chunk&7.
        const int grow = lane >> 3;      // 0..3
        const int gcol = lane & 7;       // 16B unit within a 128B row
        unsigned lds_j[4];
        unsigned long long ga_j[4];
        const unsigned ring_base = (unsigned)(uintptr_t)&ring[wave][0][0];
#pragma unroll
        for (int j = 0; j < 4; ++j) {
            const int row = j * 4 + grow;
            lds_j[j] = ring_base + (unsigned)(row * (PITCHF * 4) + gcol * 16);
            ga_j[j]  = (unsigned long long)(uintptr_t)Aop +
                       ((size_t)(row0 + row) * (size_t)K + (size_t)kb) * 4ull +
                       (unsigned long long)(gcol * 16);
        }

        auto issue = [&](int g) {
            const unsigned so = (unsigned)((g & (DEPTH - 1)) * (SLOTF * 4));
            const unsigned long long ko = (unsigned long long)g * 128ull;
#pragma unroll
            for (int j = 0; j < 4; ++j) {
                unsigned dst = lds_j[j] + so;
                unsigned long long src = ga_j[j] + ko;
                asm volatile("global_load_async_to_lds_b128 %0, %1, off"
                             :: "v"(dst), "v"(src) : "memory");
            }
        };

        // prologue: fill up to 3 groups
        for (int g = 0; g < 3 && g < nIt; ++g) issue(g);

        for (int it = 0; it < nIt; ++it) {
            if (it + 3 < nIt) issue(it + 3);

            const int rem = nIt - 1 - it;   // groups still outstanding past it
            if (rem >= 3)      asm volatile("s_wait_asynccnt 0xc" ::: "memory");
            else if (rem == 2) asm volatile("s_wait_asynccnt 0x8" ::: "memory");
            else if (rem == 1) asm volatile("s_wait_asynccnt 0x4" ::: "memory");
            else               asm volatile("s_wait_asynccnt 0x0" ::: "memory");

            // consume slot from LDS (ds_load_b128 x4), cvt to bf16 layout
            const float* rp = &ring[wave][it & (DEPTH - 1)][l * PITCHF + half * 8];
            float4 f0 = *(const float4*)(rp);
            float4 f1 = *(const float4*)(rp + 4);
            float4 f2 = *(const float4*)(rp + 16);
            float4 f3 = *(const float4*)(rp + 20);
            bf16x16 a = pack_bf16(f0, f1, f2, f3);

            const bool has_next = (it + 1 < nIt);
            if (has_next) {
                const int kn = kb + (it + 1) * 32;
#pragma unroll
                for (int t = 0; t < NT; ++t)
                    b_nxt[t] = load_bf16_row(Bt, (size_t)K, t * 16 + l, kn, half);
            }

#pragma unroll
            for (int t = 0; t < NT; ++t)
                acc[t] = wmma_bf16(a, b_cur[t], acc[t]);

            if (has_next) {
#pragma unroll
                for (int t = 0; t < NT; ++t) b_cur[t] = b_nxt[t];
            }
        }
    } else {
        // ---------------- bf16-A direct register path (L2-resident) ---------
        bf16x16 a_cur, a_nxt;
        a_cur = load_bf16_row((const __bf16*)Aop, (size_t)K, row0 + l, kb, half);
        for (int it = 0; it < nIt; ++it) {
            const bool has_next = (it + 1 < nIt);
            if (has_next) {
                const int kn = kb + (it + 1) * 32;
                a_nxt = load_bf16_row((const __bf16*)Aop, (size_t)K, row0 + l, kn, half);
#pragma unroll
                for (int t = 0; t < NT; ++t)
                    b_nxt[t] = load_bf16_row(Bt, (size_t)K, t * 16 + l, kn, half);
            }
#pragma unroll
            for (int t = 0; t < NT; ++t)
                acc[t] = wmma_bf16(a_cur, b_cur[t], acc[t]);
            if (has_next) {
                a_cur = a_nxt;
#pragma unroll
                for (int t = 0; t < NT; ++t) b_cur[t] = b_nxt[t];
            }
        }
    }

    // ---- split-K reduction through LDS (lane-private v8f slots) ----
    __shared__ v8f red[SPLIT * NT * 32];
#pragma unroll
    for (int t = 0; t < NT; ++t)
        red[(wave * NT + t) * 32 + lane] = acc[t];
    __syncthreads();

    for (int t = wave; t < NT; t += SPLIT) {
        v8f r = red[t * 32 + lane];
#pragma unroll
        for (int s = 1; s < SPLIT; ++s) r += red[(s * NT + t) * 32 + lane];

        if constexpr (TROUT) {
            bf16x8 v;
#pragma unroll
            for (int rr = 0; rr < 8; ++rr) {
                float x = r[rr];
                if (RELU) x = fmaxf(x, 0.0f);
                v[rr] = (__bf16)x;
            }
            *(bf16x8*)(Cout + (size_t)(t * 16 + l) * (size_t)M +
                       (size_t)row0 + (size_t)half * 8) = v;
        } else {
            const int Ntot = NT * 16;
#pragma unroll
            for (int rr = 0; rr < 8; ++rr) {
                float x = r[rr];
                if (RELU) x = fmaxf(x, 0.0f);
                Cout[(size_t)(row0 + half * 8 + rr) * (size_t)Ntot + t * 16 + l] =
                    (__bf16)x;
            }
        }
    }
}

// ---------------------------------------------------------------------------
// out = sigmoid(Z @ Z^T), Z bf16 [N][32] row-major; K=32 -> single WMMA/tile.
// grid = (8, N/16), block = 256 (8 waves); wave handles 8 column tiles.
// Store-bound: 64B contiguous f32 stores per half-wave row.
// ---------------------------------------------------------------------------
__global__ __launch_bounds__(256) void zzt_sigmoid_k(const __bf16* __restrict__ Z,
                                                     float* __restrict__ out,
                                                     int N) {
    const int lane = threadIdx.x & 31;
    const int half = lane >> 4;
    const int l    = lane & 15;
    const int wave = threadIdx.x >> 5;
    const int row0 = blockIdx.y * 16;

    bf16x16 a = load_bf16_row(Z, (size_t)GAE_LAT, row0 + l, 0, half);

#pragma unroll
    for (int i = 0; i < 8; ++i) {
        const int c0 = (blockIdx.x * 64 + wave * 8 + i) * 16;
        bf16x16 b = load_bf16_row(Z, (size_t)GAE_LAT, c0 + l, 0, half);
        v8f acc = zero8();
        acc = wmma_bf16(a, b, acc);
#pragma unroll
        for (int r = 0; r < 8; ++r) {
            float x = acc[r];
            float s = __builtin_amdgcn_rcpf(1.0f + __expf(-x));
            out[(size_t)(row0 + half * 8 + r) * (size_t)N + c0 + l] = s;
        }
    }
}

// Tiny fp32 [R][C] -> bf16 transposed [C][R] (for W1, W2).
__global__ void cvt_transpose_k(const float* __restrict__ src,
                                __bf16* __restrict__ dst, int R, int C) {
    int i = blockIdx.x * blockDim.x + threadIdx.x;
    if (i < R * C) {
        int r = i / C, c = i % C;
        dst[(size_t)c * R + r] = (__bf16)src[i];
    }
}

// ---------------------------------------------------------------------------
extern "C" void kernel_launch(void* const* d_in, const int* in_sizes, int n_in,
                              void* d_out, int out_size, void* d_ws, size_t ws_size,
                              hipStream_t stream) {
    const float* X  = (const float*)d_in[0];  // [8192][256]
    const float* A  = (const float*)d_in[1];  // [8192][8192]
    const float* W1 = (const float*)d_in[2];  // [256][128]
    const float* W2 = (const float*)d_in[3];  // [128][32]
    float* out = (float*)d_out;               // [8192][8192]

    char* w = (char*)d_ws;
    __bf16* W1t = (__bf16*)w; w += (size_t)GAE_HID * GAE_FIN * 2;   // [128][256]
    __bf16* W2t = (__bf16*)w; w += (size_t)GAE_LAT * GAE_HID * 2;   // [32][128]
    __bf16* B1t = (__bf16*)w; w += (size_t)GAE_HID * GAE_N * 2;     // [128][8192]
    __bf16* H   = (__bf16*)w; w += (size_t)GAE_N * GAE_HID * 2;     // [8192][128]
    __bf16* C2t = (__bf16*)w; w += (size_t)GAE_LAT * GAE_N * 2;     // [32][8192]
    __bf16* Zb  = (__bf16*)w; w += (size_t)GAE_N * GAE_LAT * 2;     // [8192][32]

    // Pre-transpose weights to bf16 (B-operand wants K-contiguous [N][K]).
    cvt_transpose_k<<<(GAE_FIN * GAE_HID + 255) / 256, 256, 0, stream>>>(
        W1, W1t, GAE_FIN, GAE_HID);
    cvt_transpose_k<<<(GAE_HID * GAE_LAT + 255) / 256, 256, 0, stream>>>(
        W2, W2t, GAE_HID, GAE_LAT);

    const int strips = GAE_N / 16;   // 512 blocks, 4 split-K waves each

    // B1t = (X @ W1)^T        [M=8192, K=256, N=128]
    gemm_splitk<8, false, false, true><<<strips, 128, 0, stream>>>(
        (const void*)X, W1t, B1t, GAE_N, GAE_FIN);

    // H = relu(A @ B1)        [M=8192, K=8192, N=128]
    gemm_splitk<8, false, true, false><<<strips, 128, 0, stream>>>(
        (const void*)A, B1t, H, GAE_N, GAE_N);

    // C2t = (H @ W2)^T        [M=8192, K=128, N=32]
    gemm_splitk<2, true, false, true><<<strips, 128, 0, stream>>>(
        (const void*)H, W2t, C2t, GAE_N, GAE_HID);

    // Z = A @ C2              [M=8192, K=8192, N=32]
    gemm_splitk<2, false, false, false><<<strips, 128, 0, stream>>>(
        (const void*)A, C2t, Zb, GAE_N, GAE_N);

    // out = sigmoid(Z @ Z^T)
    zzt_sigmoid_k<<<dim3(8, GAE_N / 16), 256, 0, stream>>>(Zb, out, GAE_N);
}
